// PairwiseRelaxedWordMoverSimilarity_7980049236098
// MI455X (gfx1250) — matrix operand order repaired and checked by reading, not compiled, not run on hardware
//
#include <hip/hip_runtime.h>

typedef __attribute__((ext_vector_type(16))) _Float16 v16h;
typedef __attribute__((ext_vector_type(8)))  _Float16 v8h;
typedef __attribute__((ext_vector_type(8)))  float    v8f;
typedef __attribute__((ext_vector_type(4)))  unsigned int u32x4;
typedef __attribute__((ext_vector_type(8)))  int          i32x8;
typedef __attribute__((ext_vector_type(4)))  int          i32x4;

constexpr int Hdim = 768;   // hidden size
constexpr int Sdim = 64;    // sequence length (S1 == S2)
constexpr int Bdim = 64;    // batch (B1 == B2)
constexpr float kNeg = -1e30f;

// ---- LDS staging geometry -------------------------------------------------
constexpr int KC     = 64;              // K-chunk (halves)
constexpr int NCHUNK = Hdim / KC;       // 12
constexpr int ROWP   = KC + 8;          // padded LDS row stride in halves (144B)
constexpr unsigned PANEL = Sdim * ROWP * 2;   // bytes per staged panel = 9216
constexpr unsigned A0OFF = 0;
constexpr unsigned A1OFF = PANEL;
constexpr unsigned B0OFF = 2 * PANEL;
constexpr unsigned B1OFF = 3 * PANEL;
constexpr unsigned MOFF  = 4 * PANEL;         // masks + reduction scratch
constexpr unsigned SMEM_BYTES = MOFF + 4 * 256;   // 37888 B (< 64KB dyn-LDS cap)

// ---------------------------------------------------------------------------
// Phase 1: unit-normalize each token vector (norm clamped at 1e-8, like the
// reference) and convert to f16 for the WMMA GEMM. One block per row.
// ---------------------------------------------------------------------------
__global__ __launch_bounds__(256)
void normalize_to_f16_kernel(const float* __restrict__ x1,
                             const float* __restrict__ x2,
                             _Float16* __restrict__ n1h,
                             _Float16* __restrict__ n2h) {
    __shared__ float red[256];
    const int row = blockIdx.x;                 // 0 .. 2*B*S-1
    const float* src;
    _Float16* dst;
    if (row < Bdim * Sdim) {
        src = x1 + (size_t)row * Hdim;
        dst = n1h + (size_t)row * Hdim;
    } else {
        const int r = row - Bdim * Sdim;
        src = x2 + (size_t)r * Hdim;
        dst = n2h + (size_t)r * Hdim;
    }
    const int tid = threadIdx.x;
    const float v0 = src[tid];
    const float v1 = src[tid + 256];
    const float v2 = src[tid + 512];
    red[tid] = v0 * v0 + v1 * v1 + v2 * v2;
    __syncthreads();
    #pragma unroll
    for (int off = 128; off > 0; off >>= 1) {
        if (tid < off) red[tid] += red[tid + off];
        __syncthreads();
    }
    const float inv = 1.0f / fmaxf(sqrtf(red[0]), 1e-8f);
    dst[tid]       = (_Float16)(v0 * inv);
    dst[tid + 256] = (_Float16)(v1 * inv);
    dst[tid + 512] = (_Float16)(v2 * inv);
}

// ---------------------------------------------------------------------------
// TDM: stage one 64 x KC f16 panel (row stride Hdim halves in memory) into LDS
// at lds_off, with TDM-side padding of 4 DWORDs after every 32-DWORD row so
// the LDS image has a 144-byte row stride (bank-conflict-free ds_load_b128).
// Descriptor per CDNA5 ISA (2-D: groups 2,3 zero).
// This toolchain exposes the 6-arg builtin form.
// ---------------------------------------------------------------------------
__device__ __forceinline__ void tdm_load_panel(const _Float16* gsrc, unsigned lds_off) {
    const unsigned long long ga = (unsigned long long)(uintptr_t)gsrc;
    u32x4 g0;
    g0[0] = 1u;                                        // count=1 (valid), user mode
    g0[1] = lds_off;                                   // lds_addr [63:32]
    g0[2] = (unsigned)(ga & 0xffffffffu);              // global_addr [95:64]
    g0[3] = (unsigned)((ga >> 32) & 0x01ffffffu)       // global_addr [120:96]
          | (2u << 30);                                // type=2 ("image") [127:126]
    i32x8 g1;
    g1[0] = (1 << 16)      // data_size = 2 bytes
          | (1 << 20)      // pad_enable
          | (4 << 22)      // pad_interval code 4 -> every 32 DWORDs (= KC*2 = 128B row)
          | (3 << 25);     // pad_amount  code 3 -> 4 DWORDs (16B) of padding
    g1[1] = (Hdim & 0xffff) << 16;   // tensor_dim0 low16  (768, never OOB)
    g1[2] = (Sdim & 0xffff) << 16;   // tensor_dim0 hi=0 | tensor_dim1 low16 (64)
    g1[3] = (KC   & 0xffff) << 16;   // tensor_dim1 hi=0 | tile_dim0 = KC
    g1[4] = Sdim;                    // tile_dim1 = 64, tile_dim2 = 0
    g1[5] = Hdim;                    // tensor_dim0_stride low32 = 768 elements
    g1[6] = 0;
    g1[7] = 0;
    const i32x4 z4 = {0, 0, 0, 0};             // groups 2/3 unused (<=2-D tensor)
    const i32x8 z8 = {0, 0, 0, 0, 0, 0, 0, 0}; // trailing group (6-arg form)
    __builtin_amdgcn_tensor_load_to_lds(g0, g1, z4, z4, z8, 0);
}

// ---------------------------------------------------------------------------
// Phase 2: one workgroup (8 wave32) per (a,b) pair.
//   TDM double-buffers 64xKC panels of n1h[a] / n2h[b] into LDS; all 8 waves
//   run v_wmma_f32_16x16x32_f16 from ds_load_b128 fragments; epilogue does the
//   masked row/col maxes + masked means in LDS (sim reuses the A panels).
// Fragments use contiguous k = 16*(lane/16) + s, the same order for A and B.
// ---------------------------------------------------------------------------
__global__ __launch_bounds__(256)
void wmvs_pair_kernel(const _Float16* __restrict__ n1h,
                      const _Float16* __restrict__ n2h,
                      const float* __restrict__ mask1,
                      const float* __restrict__ mask2,
                      float* __restrict__ out) {
    extern __shared__ char smem[];
    float* m1s  = (float*)(smem + MOFF);
    float* m2s  = m1s + 64;
    float* red1 = m1s + 128;
    float* red2 = m1s + 192;

    const int a   = blockIdx.x >> 6;
    const int b   = blockIdx.x & 63;
    const int tid = threadIdx.x;

    if (tid < Sdim)           m1s[tid] = mask1[a * Sdim + tid];
    else if (tid < 2 * Sdim)  m2s[tid - Sdim] = mask2[b * Sdim + (tid - Sdim)];

    const int wave = tid >> 5;
    const int lane = tid & 31;
    const int m0   = (wave >> 1) * 16;     // output tile row base (S1)
    const int t0   = (wave & 1) * 32;      // output tile col base (S2), 2 tiles/wave
    const int lm   = lane & 15;
    const int lhi  = lane >> 4;

    const _Float16* gA = n1h + (size_t)(a * Sdim) * Hdim;
    const _Float16* gB = n2h + (size_t)(b * Sdim) * Hdim;

    // ---- prologue: stage chunk 0 --------------------------------------------
    if (wave == 0) {
        tdm_load_panel(gA, A0OFF);
        __builtin_amdgcn_s_wait_tensorcnt(0);
    } else if (wave == 1) {
        tdm_load_panel(gB, B0OFF);
        __builtin_amdgcn_s_wait_tensorcnt(0);
    }
    __syncthreads();

    v8f acc0 = {};
    v8f acc1 = {};
    union Frag { v16h v; v8h h[2]; };

    for (int c = 0; c < NCHUNK; ++c) {
        // prefetch next chunk into the other buffer (overlaps WMMA below);
        // the barrier that ended iteration c-1 guarantees its consumers left.
        if (c + 1 < NCHUNK) {
            const unsigned pa = ((c + 1) & 1) ? A1OFF : A0OFF;
            const unsigned pb = ((c + 1) & 1) ? B1OFF : B0OFF;
            if (wave == 0)      tdm_load_panel(gA + (size_t)(c + 1) * KC, pa);
            else if (wave == 1) tdm_load_panel(gB + (size_t)(c + 1) * KC, pb);
        }

        const _Float16* Ab = (const _Float16*)(smem + ((c & 1) ? A1OFF : A0OFF));
        const _Float16* Bb = (const _Float16*)(smem + ((c & 1) ? B1OFF : B0OFF));
        const _Float16* pA  = Ab + (m0 + lm) * ROWP + lhi * 16;
        const _Float16* pB0 = Bb + (t0 + lm) * ROWP + lhi * 16;
        const _Float16* pB1 = pB0 + 16 * ROWP;

        #pragma unroll
        for (int kk = 0; kk < KC; kk += 32) {
            Frag fa, fb0, fb1;
            fa.h[0]  = *(const v8h*)(pA + kk);
            fa.h[1]  = *(const v8h*)(pA + kk + 8);
            fb0.h[0] = *(const v8h*)(pB0 + kk);
            fb0.h[1] = *(const v8h*)(pB0 + kk + 8);
            fb1.h[0] = *(const v8h*)(pB1 + kk);
            fb1.h[1] = *(const v8h*)(pB1 + kk + 8);
            acc0 = __builtin_amdgcn_wmma_f32_16x16x32_f16(
                false, fa.v, false, fb0.v, (short)0, acc0, false, false);
            acc1 = __builtin_amdgcn_wmma_f32_16x16x32_f16(
                false, fa.v, false, fb1.v, (short)0, acc1, false, false);
        }

        // issuing waves drain the in-flight DMA, then everyone syncs so the
        // freshly staged buffer is visible (and this buffer is reusable).
        if (wave < 2) __builtin_amdgcn_s_wait_tensorcnt(0);
        __syncthreads();
    }

    // ---- epilogue: sim buffer reuses the A panels (done with staging) -------
    constexpr int kPitch = Sdim + 1;                 // 65 floats, conflict-free
    float* sim = (float*)smem;                        // 64*65*4 = 16640 <= 2*PANEL

    #pragma unroll
    for (int r = 0; r < 8; ++r) {                    // D: (M = r + 8*lhi, N = lm)
        const int srow = m0 + r + lhi * 8;
        const int tcol = t0 + lm;
        sim[srow * kPitch + tcol]      = acc0[r];
        sim[srow * kPitch + tcol + 16] = acc1[r];
    }
    __syncthreads();

    if (tid < Sdim) {
        const int s = tid;
        float mx = kNeg;
        for (int t = 0; t < Sdim; ++t) {
            const float v = (m2s[t] > 0.0f) ? sim[s * kPitch + t] : kNeg;
            mx = fmaxf(mx, v);
        }
        red1[s] = (m1s[s] > 0.0f) ? mx : 0.0f;
    } else if (tid < 2 * Sdim) {
        const int t = tid - Sdim;
        float mx = kNeg;
        for (int s = 0; s < Sdim; ++s) {
            const float v = (m1s[s] > 0.0f) ? sim[s * kPitch + t] : kNeg;
            mx = fmaxf(mx, v);
        }
        red2[t] = (m2s[t] > 0.0f) ? mx : 0.0f;
    }
    __syncthreads();

    if (tid == 0) {
        float s1 = 0.0f, s2 = 0.0f, c1 = 0.0f, c2 = 0.0f;
        for (int i = 0; i < Sdim; ++i) {
            s1 += red1[i];
            s2 += red2[i];
            c1 += (m1s[i] > 0.0f) ? 1.0f : 0.0f;
            c2 += (m2s[i] > 0.0f) ? 1.0f : 0.0f;
        }
        out[blockIdx.x] = 0.5f * (s1 / c1 + s2 / c2);
    }
}

// ---------------------------------------------------------------------------
extern "C" void kernel_launch(void* const* d_in, const int* in_sizes, int n_in,
                              void* d_out, int out_size, void* d_ws, size_t ws_size,
                              hipStream_t stream) {
    (void)in_sizes; (void)n_in; (void)out_size; (void)ws_size;
    const float* x1    = (const float*)d_in[0];
    const float* mask1 = (const float*)d_in[1];
    const float* x2    = (const float*)d_in[2];
    const float* mask2 = (const float*)d_in[3];
    float* out = (float*)d_out;

    _Float16* n1h = (_Float16*)d_ws;                                  // 6.29 MB
    _Float16* n2h = n1h + (size_t)Bdim * Sdim * Hdim;                 // 6.29 MB

    normalize_to_f16_kernel<<<2 * Bdim * Sdim, 256, 0, stream>>>(x1, x2, n1h, n2h);
    wmvs_pair_kernel<<<Bdim * Bdim, 256, SMEM_BYTES, stream>>>(n1h, n2h, mask1, mask2, out);
}
